// ClusterMemory_47923245088805
// MI455X (gfx1250) — matrix operand order repaired and checked by reading, not compiled, not run on hardware
//
#include <hip/hip_runtime.h>
#include <hip/hip_bf16.h>
#include <math.h>

// ---------------------------------------------------------------------------
// Problem constants (from the reference)
// ---------------------------------------------------------------------------
#define B      256
#define D      768
#define NR     4096
#define NI     4096
#define A_TOT  8192
#define KSTEPS (D / 32)          // 24 K-steps of 32
#define INV_TEMP 20.0f           // 1 / 0.05
#define MOM    0.2f
#define LOG_HALF (-0.69314718055994530942f)

typedef __attribute__((ext_vector_type(16))) __bf16 v16bf;
typedef __attribute__((ext_vector_type(8)))  __bf16 v8bf;
typedef __attribute__((ext_vector_type(8)))  float  v8f;
typedef __attribute__((ext_vector_type(4)))  int    v4i;

typedef v4i __attribute__((address_space(1))) as1_v4i;   // global
typedef v4i __attribute__((address_space(3))) as3_v4i;   // LDS

union FragBF16 { v16bf v; v8bf h[2]; };

// ---------------------------------------------------------------------------
// CDNA5 async global->LDS staging (ASYNCcnt-tracked DMA) with fallback.
// Probe-learned prototype: (AS1 v4i* gsrc, AS3 v4i* ldst, imm off, imm cpol)
// ---------------------------------------------------------------------------
#if defined(__gfx1250__) && __has_builtin(__builtin_amdgcn_global_load_async_to_lds_b128)
#define HAVE_ASYNC_LDS 1
#else
#define HAVE_ASYNC_LDS 0
#endif

__device__ __forceinline__ void stage16(const __bf16* g, __bf16* l) {
#if HAVE_ASYNC_LDS
  __builtin_amdgcn_global_load_async_to_lds_b128(
      (as1_v4i*)g, (as3_v4i*)l, 0, 0);
#else
  *reinterpret_cast<v8bf*>(l) = *reinterpret_cast<const v8bf*>(g);  // ds_store_b128
#endif
}

__device__ __forceinline__ void stage_wait() {
#if HAVE_ASYNC_LDS
#if __has_builtin(__builtin_amdgcn_s_wait_asynccnt)
  __builtin_amdgcn_s_wait_asynccnt(0);
#else
  asm volatile("s_wait_asynccnt 0x0" ::: "memory");
#endif
#endif
}

// ---------------------------------------------------------------------------
// Kernel 1: row-normalize inputs -> x_f32 (for the sequential update) and
// x_bf16 (GEMM A operand). One block per row, 256 threads x 3 elements.
// ---------------------------------------------------------------------------
__global__ __launch_bounds__(256)
void k_normalize(const float* __restrict__ in, float* __restrict__ x,
                 __bf16* __restrict__ xb) {
  const int row = blockIdx.x;
  const int tid = threadIdx.x;
  const float* r = in + (size_t)row * D;
  float v0 = r[tid], v1 = r[tid + 256], v2 = r[tid + 512];
  __shared__ float red[256];
  red[tid] = v0 * v0 + v1 * v1 + v2 * v2;
  __syncthreads();
  for (int s = 128; s > 0; s >>= 1) {
    if (tid < s) red[tid] += red[tid + s];
    __syncthreads();
  }
  const float inv = 1.0f / sqrtf(red[0]);
  float x0 = v0 * inv, x1 = v1 * inv, x2 = v2 * inv;
  float* xr = x + (size_t)row * D;
  __bf16* xbr = xb + (size_t)row * D;
  xr[tid] = x0;        xr[tid + 256] = x1;        xr[tid + 512] = x2;
  xbr[tid] = (__bf16)x0; xbr[tid + 256] = (__bf16)x1; xbr[tid + 512] = (__bf16)x2;
}

// ---------------------------------------------------------------------------
// Kernel 2: fused convert+transpose of concat(features_rgb, features_ir)
// (row-major N x K, fp32) into Bt (K-major: Bt[k][n], bf16) so B fragments
// are contiguous-N loads. 32x32 tiles via LDS; coalesced both directions.
// ---------------------------------------------------------------------------
__global__ __launch_bounds__(256)
void k_transpose(const float* __restrict__ frgb, const float* __restrict__ fir,
                 __bf16* __restrict__ Bt) {
  __shared__ __bf16 tile[32][33];
  const int tx = threadIdx.x & 31;
  const int ty = threadIdx.x >> 5;
  const int nBase = blockIdx.x * 32;
  const int kBase = blockIdx.y * 32;
#pragma unroll
  for (int r = 0; r < 4; ++r) {
    const int n = nBase + ty + 8 * r;
    const int k = kBase + tx;
    const float v = (n < NR) ? frgb[(size_t)n * D + k]
                             : fir[(size_t)(n - NR) * D + k];
    tile[ty + 8 * r][tx] = (__bf16)v;
  }
  __syncthreads();
#pragma unroll
  for (int r = 0; r < 4; ++r) {
    const int k = kBase + ty + 8 * r;
    const int n = nBase + tx;
    Bt[(size_t)k * A_TOT + n] = tile[tx][ty + 8 * r];
  }
}

// ---------------------------------------------------------------------------
// Kernel 3: WMMA GEMM with double-buffered async LDS staging of B.
//   S[256 x 8192] = x_bf16 (256x768) @ Bt (768x8192), f32 accumulate.
// One block per 64-wide N group (128 blocks); block covers full M=256:
// 8 waves x 2 M-tiles, 8 WMMAs per wave per K-step.  B chunk (32K x 64N,
// 4 KB) staged once per block per K-step via ASYNCcnt DMA, overlapped with
// compute on the other buffer; one barrier per K-step.
// LDS row stride 72 bf16 (144 B): rows k and k+8, read at once by the two
// lane halves for one B-fragment element, start 32 banks apart.
// Fragment layouts per CDNA5 ISA 7.12.2 (wave32):
//   A: lane = M%16; ks = lane>=16 ? {8..15,24..31} : {0..7,16..23}
//   B: lane = N%16; ks = lane>=16 ? {8..15,24..31} : {0..7,16..23}
//   C: lane = N%16; vgpr v = M (lanes>=16: M+8)
// ---------------------------------------------------------------------------
#define LDS_STRIDE 72
__global__ __launch_bounds__(256)
void k_gemm(const __bf16* __restrict__ Xb, const __bf16* __restrict__ Bt,
            float* __restrict__ S) {
  __shared__ __align__(16) __bf16 ldsB[2][32 * LDS_STRIDE];
  const int tid  = threadIdx.x;
  const int lane = tid & 31;
  const int wave = tid >> 5;
  const int nBase = blockIdx.x * 64;     // 128 N groups
  const int l15 = lane & 15;
  const int lhi = lane >> 4;
  const int mBaseW = wave * 32;          // wave owns M-tiles 2w, 2w+1

  // Staging coords: 2048 bf16 per chunk, 8 per thread (one 16B transfer)
  const int srow = tid >> 3;             // 0..31 (k within chunk)
  const int scol = (tid & 7) * 8;        // 0..56 (n within group)
  const __bf16* gstage0 = Bt + (size_t)srow * A_TOT + nBase + scol;
  __bf16* lstage0 = &ldsB[0][srow * LDS_STRIDE + scol];
  __bf16* lstage1 = &ldsB[1][srow * LDS_STRIDE + scol];

  v8f acc[2][4] = {};

  // Prologue: stage K-step 0 into buffer 0
  stage16(gstage0, lstage0);
  stage_wait();
  __syncthreads();

  int buf = 0;
  for (int ks = 0; ks < KSTEPS; ++ks) {
    if (ks + 1 < KSTEPS)
      stage16(gstage0 + (size_t)(ks + 1) * 32 * A_TOT,
              (buf == 0) ? lstage1 : lstage0);

    // B fragments: lane column n = l15; elems k = table order (lhi*8+{0..7,16..23})
    const __bf16* lb = &ldsB[buf][0];
    FragBF16 fb[4];
#pragma unroll
    for (int t = 0; t < 4; ++t) {
#pragma unroll
      for (int s = 0; s < 16; ++s) {
        const int kk = (s < 8) ? (lhi * 8 + s) : (16 + lhi * 8 + (s - 8));
        fb[t].v[s] = lb[kk * LDS_STRIDE + t * 16 + l15];
      }
    }
#pragma unroll
    for (int mt = 0; mt < 2; ++mt) {
      const __bf16* arow =
          Xb + (size_t)(mBaseW + mt * 16 + l15) * D + ks * 32 + lhi * 8;
      FragBF16 fa;
      fa.h[0] = *reinterpret_cast<const v8bf*>(arow);
      fa.h[1] = *reinterpret_cast<const v8bf*>(arow + 16);
#pragma unroll
      for (int t = 0; t < 4; ++t)
        acc[mt][t] = __builtin_amdgcn_wmma_f32_16x16x32_bf16(
            false, fa.v, false, fb[t].v, (short)0, acc[mt][t], false, false);
    }

    if (ks + 1 < KSTEPS) stage_wait();   // own DMA done (overlapped compute)
    __syncthreads();                     // all waves' chunk visible / reads done
    buf ^= 1;
  }

#pragma unroll
  for (int mt = 0; mt < 2; ++mt) {
#pragma unroll
    for (int v = 0; v < 8; ++v) {
      const int row = mBaseW + mt * 16 + v + (lhi << 3);
      float* srowp = S + (size_t)row * A_TOT + nBase + l15;
#pragma unroll
      for (int t = 0; t < 4; ++t) srowp[t * 16] = acc[mt][t][v];
    }
  }
}

// ---------------------------------------------------------------------------
// Kernel 4: per-row loss statistics.  For each half h (rgb / ir):
//   z = S/TEMP; LSE = max + log(sum exp(z-max)); Dh = sum ct*z; Ch = sum ct
//   loss_yc_row = -(D0 - LSE0*C0)                         (rgb only)
//   loss_y_row  = sum_h -(Dh - LSEh*Ch + log(0.5)*Ch)
// ---------------------------------------------------------------------------
__global__ __launch_bounds__(256)
void k_rowloss(const float* __restrict__ S, const float* __restrict__ ct,
               float* __restrict__ rowYC, float* __restrict__ rowY) {
  const int i = blockIdx.x;
  const int tid = threadIdx.x;
  __shared__ float r0[256], r1[256], r2[256];
  float lyc = 0.0f, ly = 0.0f;

  for (int h = 0; h < 2; ++h) {
    const float* s = S + (size_t)i * A_TOT + h * NR;
    const float* c = ct + (size_t)i * A_TOT + h * NR;

    float mx = -3.4e38f;
#pragma unroll
    for (int e = 0; e < 16; ++e)
      mx = fmaxf(mx, s[tid + e * 256] * INV_TEMP);
    r0[tid] = mx;
    __syncthreads();
    for (int st = 128; st > 0; st >>= 1) {
      if (tid < st) r0[tid] = fmaxf(r0[tid], r0[tid + st]);
      __syncthreads();
    }
    mx = r0[0];
    __syncthreads();

    float se = 0.0f, Dv = 0.0f, Cv = 0.0f;
#pragma unroll
    for (int e = 0; e < 16; ++e) {
      const int idx = tid + e * 256;
      const float z = s[idx] * INV_TEMP;
      const float cv = c[idx];
      se += expf(z - mx);
      Dv += cv * z;
      Cv += cv;
    }
    r0[tid] = se; r1[tid] = Dv; r2[tid] = Cv;
    __syncthreads();
    for (int st = 128; st > 0; st >>= 1) {
      if (tid < st) {
        r0[tid] += r0[tid + st];
        r1[tid] += r1[tid + st];
        r2[tid] += r2[tid + st];
      }
      __syncthreads();
    }
    const float LSE = mx + logf(r0[0]);
    const float Dh = r1[0], Ch = r2[0];
    if (h == 0) lyc = -(Dh - LSE * Ch);
    ly += -(Dh - LSE * Ch + LOG_HALF * Ch);
    __syncthreads();
  }
  if (tid == 0) { rowYC[i] = lyc; rowY[i] = ly; }
}

// ---------------------------------------------------------------------------
// Kernel 5: mean over the 256 rows -> d_out[0], d_out[1]
// ---------------------------------------------------------------------------
__global__ __launch_bounds__(256)
void k_meanloss(const float* __restrict__ rowYC, const float* __restrict__ rowY,
                float* __restrict__ out) {
  __shared__ float r0[256], r1[256];
  const int tid = threadIdx.x;
  r0[tid] = rowYC[tid];
  r1[tid] = rowY[tid];
  __syncthreads();
  for (int s = 128; s > 0; s >>= 1) {
    if (tid < s) { r0[tid] += r0[tid + s]; r1[tid] += r1[tid + s]; }
    __syncthreads();
  }
  if (tid == 0) {
    out[0] = r0[0] * (1.0f / B);
    out[1] = r1[0] * (1.0f / B);
  }
}

// ---------------------------------------------------------------------------
// Kernel 6: bulk copy features_all -> output (float4, fully coalesced)
// ---------------------------------------------------------------------------
__global__ __launch_bounds__(256)
void k_copy4(const float4* __restrict__ src, float4* __restrict__ dst) {
  const size_t i = (size_t)blockIdx.x * 256 + threadIdx.x;
  dst[i] = src[i];
}

// ---------------------------------------------------------------------------
// Kernel 7: faithful sequential momentum scatter + per-row L2 renorm.
// mask = (pids_ir == target[i])  =>  row j = target[i]-4096 of F[:4096].
// Single block, 256 threads x 3 elems; branch is block-uniform.
// ---------------------------------------------------------------------------
__global__ __launch_bounds__(256)
void k_update(const float* __restrict__ x, const int* __restrict__ targets,
              const float* __restrict__ ct, float* __restrict__ F) {
  const int tid = threadIdx.x;
  __shared__ float red[256];
  for (int i = 0; i < B; ++i) {
    const int t = targets[i];
    if (t >= NR) {                       // uniform branch
      const int j = t - NR;
      const float yci = ct[(size_t)i * A_TOT + t];
      const float m = MOM / fmaxf(yci, MOM);
      float* frow = F + (size_t)j * D;
      const float* xrow = x + (size_t)i * D;
      const float u0 = m * frow[tid]       + (1.0f - m) * xrow[tid];
      const float u1 = m * frow[tid + 256] + (1.0f - m) * xrow[tid + 256];
      const float u2 = m * frow[tid + 512] + (1.0f - m) * xrow[tid + 512];
      red[tid] = u0 * u0 + u1 * u1 + u2 * u2;
      __syncthreads();
      for (int s = 128; s > 0; s >>= 1) {
        if (tid < s) red[tid] += red[tid + s];
        __syncthreads();
      }
      const float inv = 1.0f / sqrtf(red[0]);
      __syncthreads();
      frow[tid]       = u0 * inv;
      frow[tid + 256] = u1 * inv;
      frow[tid + 512] = u2 * inv;
      __syncthreads();
    }
  }
}

// ---------------------------------------------------------------------------
// Host-side orchestration
// ---------------------------------------------------------------------------
extern "C" void kernel_launch(void* const* d_in, const int* in_sizes, int n_in,
                              void* d_out, int out_size, void* d_ws, size_t ws_size,
                              hipStream_t stream) {
  (void)in_sizes; (void)n_in; (void)out_size; (void)ws_size;
  const float* inputs = (const float*)d_in[0];   // 256 x 768
  const int*   targets = (const int*)d_in[1];    // 256
  const float* ct = (const float*)d_in[2];       // 256 x 8192
  const float* frgb = (const float*)d_in[3];     // 4096 x 768
  const float* fir  = (const float*)d_in[4];     // 4096 x 768
  const float* fall = (const float*)d_in[5];     // 8192 x 768
  float* out = (float*)d_out;                    // [loss_yc, loss_y, 8192x768]

  // Workspace carve-out (all offsets 256B-aligned):
  char* ws = (char*)d_ws;
  float*  x_f32 = (float*)(ws + 0);                       //  768 KB
  __bf16* x_bf  = (__bf16*)(ws + 786432);                 //  384 KB
  __bf16* Bt    = (__bf16*)(ws + 1179648);                // 12.0 MB (768 x 8192)
  float*  S     = (float*)(ws + 13762560);                //  8.0 MB (256 x 8192)
  float*  rowYC = (float*)(ws + 22151168);                //  1 KB
  float*  rowY  = (float*)(ws + 22152192);                //  1 KB

  // 1) normalize inputs -> x (f32) and A operand (bf16)
  k_normalize<<<B, 256, 0, stream>>>(inputs, x_f32, x_bf);

  // 2) convert+transpose features into K-major bf16 B operand
  k_transpose<<<dim3(A_TOT / 32, D / 32), 256, 0, stream>>>(frgb, fir, Bt);

  // 3) WMMA GEMM: S = x @ F^T (128 blocks; LDS-staged, double-buffered B)
  k_gemm<<<A_TOT / 64, 256, 0, stream>>>(x_bf, Bt, S);

  // 4) per-row softmax statistics -> per-row losses
  k_rowloss<<<B, 256, 0, stream>>>(S, ct, rowYC, rowY);

  // 5) mean losses -> out[0], out[1]
  k_meanloss<<<1, 256, 0, stream>>>(rowYC, rowY, out);

  // 6) copy features_all into the output, then 7) sequential update in place
  k_copy4<<<(A_TOT * D) / 4 / 256, 256, 0, stream>>>(
      (const float4*)fall, (float4*)(out + 2));
  k_update<<<1, 256, 0, stream>>>(x_f32, targets, ct, out + 2);
}
